// StatefulStreamingSemanticEncoder_60095182405752
// MI455X (gfx1250) — compile-verified, matrix operand order checked by reading, not run on hardware
//
#include <hip/hip_runtime.h>
#include <hip/hip_bf16.h>

// ---------------------------------------------------------------------------
// Types for CDNA5 WMMA
// ---------------------------------------------------------------------------
typedef __attribute__((ext_vector_type(16))) __bf16 v16bf;
typedef __attribute__((ext_vector_type(8)))  __bf16 v8bf;
typedef __attribute__((ext_vector_type(8)))  float  v8f;

#define MODE_PLAIN     0
#define MODE_GELU_BF16 1
#define MODE_RES       2

// Fragment element layout (CDNA5 ISA 7.12.2, 16-bit A/B 16x32):
//   elements 0..7  = K at p[off + 0..7]
//   elements 8..15 = K at p[off + 16..23]   (off = 0 lanes 0-15, 8 lanes 16-31)
__device__ __forceinline__ void load_frag16(const float* p, int off, float* f) {
    const float4 a = *(const float4*)(p + off);
    const float4 b = *(const float4*)(p + off + 4);
    const float4 c = *(const float4*)(p + off + 16);
    const float4 d = *(const float4*)(p + off + 20);
    f[0]=a.x; f[1]=a.y; f[2]=a.z; f[3]=a.w;
    f[4]=b.x; f[5]=b.y; f[6]=b.z; f[7]=b.w;
    f[8]=c.x; f[9]=c.y; f[10]=c.z; f[11]=c.w;
    f[12]=d.x; f[13]=d.y; f[14]=d.z; f[15]=d.w;
}
__device__ __forceinline__ v16bf frag_bf16(const __bf16* p, int off) {
    const v8bf c0 = *(const v8bf*)(p + off);
    const v8bf c1 = *(const v8bf*)(p + off + 16);
    v16bf v;
#pragma unroll
    for (int i = 0; i < 8; ++i) { v[i] = c0[i]; v[8 + i] = c1[i]; }
    return v;
}
// A-fragment builders; fc carries the shared RMSNorm column weights (f32 path)
__device__ __forceinline__ v16bf mk_afrag(const float* p, int off,
                                          const float* fc, bool use_cs) {
    float f[16];
    load_frag16(p, off, f);
    if (use_cs) {
#pragma unroll
        for (int i = 0; i < 16; ++i) f[i] *= fc[i];
    }
    v16bf v;
#pragma unroll
    for (int i = 0; i < 16; ++i) v[i] = (__bf16)f[i];
    return v;
}
__device__ __forceinline__ v16bf mk_afrag(const __bf16* p, int off,
                                          const float*, bool) {
    return frag_bf16(p, off);
}

// ---------------------------------------------------------------------------
// WMMA GEMM:  out[m][n] = epilogue( rowscale[m] * (sum_k A'[m][k]*W[n][k]) + bias[n] )
//   with A'[m][k] = A[m][k] * colscale[k]   (rowscale/colscale = fused RMSNorm;
//   the row factor commutes with the K-sum so it is applied to the f32
//   accumulator in the epilogue, keeping the hot loop lean)
//   A: (M x K) row-major, stride lda (float or bf16)
//   W: (N x K) row-major bf16 (pre-converted; K contiguous = WMMA B columns)
// 256 threads = 8 waves arranged 4(M)x2(N); each wave owns a 32x32 tile =
// 2x2 WMMA tiles, 4 independent accumulators, fragments reused twice each.
// Requires: M % 128 == 0, N % 32 == 0, K % 32 == 0 (true for every layer).
// ---------------------------------------------------------------------------
template<int MODE, typename TA>
__device__ __forceinline__ void store_tile(const v8f& acc, int m0, int n0,
                                           int lane, const float* bias,
                                           const float* rowscale,
                                           const float* res,
                                           const float* resgamma,
                                           void* out, long ldo) {
    const int col = n0 + (lane & 15);
    const float bcol = bias ? bias[col] : 0.0f;
#pragma unroll
    for (int i = 0; i < 8; ++i) {
        const int row = m0 + i + (lane >> 4) * 8;   // C/D: VGPR i -> M=i / i+8
        float v = acc[i];
        if (rowscale) v *= rowscale[row];
        v += bcol;
        if (MODE == MODE_GELU_BF16) {
            v = 0.5f * v * (1.0f + erff(v * 0.70710678118654752f)); // exact GELU
            ((__bf16*)out)[(size_t)row * ldo + col] = (__bf16)v;
        } else if (MODE == MODE_RES) {
            ((float*)out)[(size_t)row * ldo + col] =
                res[(size_t)row * ldo + col] + resgamma[col] * v;
        } else {
            ((float*)out)[(size_t)row * ldo + col] = v;
        }
    }
}

template<int MODE, typename TA>
__global__ void gemm_wmma_kernel(const TA* __restrict__ A, long lda,
                                 const __bf16* __restrict__ W,
                                 const float* __restrict__ bias,
                                 const float* __restrict__ rowscale,
                                 const float* __restrict__ colscale,
                                 const float* __restrict__ res,
                                 const float* __restrict__ resgamma,
                                 void* __restrict__ out, long ldo,
                                 int M, int N, int K) {
    const int lane = threadIdx.x & 31;
    const int wave = threadIdx.x >> 5;
    const int m0 = blockIdx.x * 128 + (wave >> 1) * 32;
    const int n0 = blockIdx.y * 64 + (wave & 1) * 32;
    if (n0 >= N) return;                       // uniform per wave: EXEC stays all-1s

    const int off = (lane >> 4) * 8;
    const TA*     Ap0 = A + (size_t)(m0 + (lane & 15)) * lda;
    const TA*     Ap1 = A + (size_t)(m0 + 16 + (lane & 15)) * lda;
    const __bf16* Wp0 = W + (size_t)(n0 + (lane & 15)) * K;
    const __bf16* Wp1 = W + (size_t)(n0 + 16 + (lane & 15)) * K;
    const bool use_cs = (colscale != nullptr);

    v8f acc00 = {}, acc01 = {}, acc10 = {}, acc11 = {};
    for (int k0 = 0; k0 < K; k0 += 32) {
        float fc[16];
        if (use_cs) load_frag16(colscale + k0, off, fc);   // shared by a0 and a1
        const v16bf a0 = mk_afrag(Ap0 + k0, off, fc, use_cs);
        const v16bf a1 = mk_afrag(Ap1 + k0, off, fc, use_cs);
        const v16bf b0 = frag_bf16(Wp0 + k0, off);
        const v16bf b1 = frag_bf16(Wp1 + k0, off);
        __builtin_prefetch(Ap0 + k0 + 128, 0, 1);          // global_prefetch_b8
        __builtin_prefetch(Ap1 + k0 + 128, 0, 1);
        acc00 = __builtin_amdgcn_wmma_f32_16x16x32_bf16(false, a0, false, b0,
                                                        (short)0, acc00, false, false);
        acc01 = __builtin_amdgcn_wmma_f32_16x16x32_bf16(false, a0, false, b1,
                                                        (short)0, acc01, false, false);
        acc10 = __builtin_amdgcn_wmma_f32_16x16x32_bf16(false, a1, false, b0,
                                                        (short)0, acc10, false, false);
        acc11 = __builtin_amdgcn_wmma_f32_16x16x32_bf16(false, a1, false, b1,
                                                        (short)0, acc11, false, false);
    }

    store_tile<MODE, TA>(acc00, m0,      n0,      lane, bias, rowscale, res, resgamma, out, ldo);
    store_tile<MODE, TA>(acc01, m0,      n0 + 16, lane, bias, rowscale, res, resgamma, out, ldo);
    store_tile<MODE, TA>(acc10, m0 + 16, n0,      lane, bias, rowscale, res, resgamma, out, ldo);
    store_tile<MODE, TA>(acc11, m0 + 16, n0 + 16, lane, bias, rowscale, res, resgamma, out, ldo);
}

// ---------------------------------------------------------------------------
// RMSNorm row factors, wave-per-row (wave32 shuffle reduction, 8 rows/block)
// ---------------------------------------------------------------------------
__global__ void rms_rinv_kernel(const float* __restrict__ x,
                                float* __restrict__ rinv, int C, long T) {
    const int  lane = threadIdx.x & 31;
    const long t = (long)blockIdx.x * 8 + (threadIdx.x >> 5);
    if (t >= T) return;
    float s = 0.0f;
    for (int c = lane; c < C; c += 32) {
        const float v = x[t * C + c];
        s += v * v;
    }
#pragma unroll
    for (int m = 16; m > 0; m >>= 1) s += __shfl_xor(s, m, 32);
    if (lane == 0) rinv[t] = rsqrtf(s / (float)C + 1e-5f);
}

// ---------------------------------------------------------------------------
// Fused: h = rms(x)*norm_w ; h = causal depthwise conv7(h, cache) ;
//        xout = x + gamma * (h + b)
// ---------------------------------------------------------------------------
__global__ void dwconv_block_kernel(const float* __restrict__ xin,
                                    const float* __restrict__ rinv,
                                    const float* __restrict__ w,
                                    const float* __restrict__ b,
                                    const float* __restrict__ cache,
                                    const float* __restrict__ normw,
                                    const float* __restrict__ gamma,
                                    float* __restrict__ xout, int C, long T) {
    const long idx = (long)blockIdx.x * blockDim.x + threadIdx.x;
    if (idx >= T * C) return;
    const int  c = (int)(idx % C);
    const long t = idx / C;
    const float nw = normw[c];
    float acc = b[c];
#pragma unroll
    for (int k = 0; k < 7; ++k) {
        const long tp = t - 6 + k;
        const float h = (tp >= 0) ? xin[tp * C + c] * rinv[tp] * nw
                                  : cache[(size_t)c * 6 + (int)(tp + 6)];
        acc += w[c * 7 + k] * h;
    }
    xout[t * C + c] = xin[t * C + c] + gamma[c] * acc;
}

// ---------------------------------------------------------------------------
// Stem conv: 1 -> 32 channels, k=7, causal (cache = last 6 input samples)
// ---------------------------------------------------------------------------
__global__ void stem_kernel(const float* __restrict__ audio,
                            const float* __restrict__ w,
                            const float* __restrict__ b,
                            const float* __restrict__ cache,
                            float* __restrict__ out, long T) {
    const long idx = (long)blockIdx.x * blockDim.x + threadIdx.x;
    if (idx >= T * 32) return;
    const int  c = (int)(idx & 31);
    const long t = idx >> 5;
    float acc = b[c];
#pragma unroll
    for (int k = 0; k < 7; ++k) {
        const long tp = t - 6 + k;
        const float v = (tp >= 0) ? audio[tp] : cache[tp + 6];
        acc += w[c * 7 + k] * v;
    }
    out[t * 32 + c] = acc;
}

// cache (C, ctx) channel-major -> ctx time-rows (ctx, C) just before the base
__global__ void fill_cache_rows_kernel(const float* __restrict__ cache,
                                       float* __restrict__ dst, int C, int ctx) {
    const int idx = blockIdx.x * blockDim.x + threadIdx.x;
    if (idx >= C * ctx) return;
    const int c = idx / ctx, j = idx % ctx;
    dst[(size_t)j * C + c] = cache[(size_t)c * ctx + j];
}

// conv weights (O, C, Kk) f32 -> (O, Kk*C) bf16 with (k,ci) -> k*C+ci so the
// strided conv becomes a GEMM over contiguous time rows (free im2col)
__global__ void reorder_w_kernel(const float* __restrict__ w,
                                 __bf16* __restrict__ o, int O, int C, int Kk) {
    const long idx = (long)blockIdx.x * blockDim.x + threadIdx.x;
    if (idx >= (long)O * C * Kk) return;
    const int  k  = (int)(idx % Kk);
    const long r  = idx / Kk;
    const int  ci = (int)(r % C);
    const int  oo = (int)(r / C);
    o[(size_t)oo * C * Kk + (size_t)k * C + ci] = (__bf16)w[idx];
}

// plain f32 -> bf16 weight conversion (layout preserved)
__global__ void f32_to_bf16_kernel(const float* __restrict__ src,
                                   __bf16* __restrict__ dst, long n) {
    const long i = (long)blockIdx.x * blockDim.x + threadIdx.x;
    if (i < n) dst[i] = (__bf16)src[i];
}

// (T=128, N=128) time-major -> (C=128, T=128) channel-major output
__global__ void transpose128_kernel(const float* __restrict__ in,
                                    float* __restrict__ out) {
    const int idx = blockIdx.x * blockDim.x + threadIdx.x;
    if (idx >= 128 * 128) return;
    const int t = idx & 127, c = idx >> 7;
    out[c * 128 + t] = in[t * 128 + c];
}

// ---------------------------------------------------------------------------
// Host orchestration
// ---------------------------------------------------------------------------
extern "C" void kernel_launch(void* const* d_in, const int* in_sizes, int n_in,
                              void* d_out, int out_size, void* d_ws, size_t ws_size,
                              hipStream_t stream) {
    (void)in_sizes; (void)n_in; (void)out_size; (void)ws_size;
    static const int DEPTH[7] = {3, 3, 3, 3, 3, 3, 8};
    static const int RAT[6]   = {2, 2, 4, 5, 5, 8};
    const int PREF = 8; // prefix rows reserved for causal cache before each base

    // ---- workspace carve-up (all offsets 256B aligned) ----
    const size_t ACT_B = 52432896;   // max (T+8)*C floats = 13,108,224 * 4
    const size_t REG_B = 106954752;  // H1 bf16 (max 104.8MB) + bf16 FFN weights
    char*  ws   = (char*)d_ws;
    float* bufA = (float*)ws;
    float* bufB = (float*)(ws + ACT_B);
    char*  reg  = ws + 2 * ACT_B;    // H1 intermediate / bf16 weight scratch
    float* rinv = (float*)(ws + 2 * ACT_B + REG_B);
    float* htmp = (float*)(ws + 2 * ACT_B + REG_B + 1638400);
    __bf16* h1  = (__bf16*)reg;
    __bf16* wbf = (__bf16*)reg;      // conv-weight scratch (H1 not live then)

    // ---- parameter walk (setup_inputs dict insertion order; None dropped) ----
    int p = 0;
    auto nf = [&]() { return (const float*)d_in[p++]; };
    const float* audio = nf();
    const float* st_w = nf(); const float* st_b = nf(); const float* st_c = nf();

    long   T   = 3200L * 128L;       // 409600
    float* cur = bufA;
    float* alt = bufB;
    int C = 32;

    { // stem: 1 -> 32
        const long n = T * 32;
        stem_kernel<<<(unsigned)((n + 255) / 256), 256, 0, stream>>>(
            audio, st_w, st_b, st_c, cur + (size_t)PREF * C, T);
    }

    for (int s = 0; s < 7; ++s) {
        C = 32 << s;
        const int C4 = 4 * C;
        // bf16 FFN-weight scratch lives right after this stage's H1 region
        __bf16* wb1 = (__bf16*)(reg + (size_t)T * C4 * 2);
        __bf16* wb2 = wb1 + (size_t)C4 * C;

        for (int j = 0; j < DEPTH[s]; ++j) {
            const float* cw  = nf(); const float* cb  = nf(); const float* cc = nf();
            const float* nw  = nf(); const float* fnw = nf();
            const float* gm  = nf(); const float* fgm = nf();
            const float* l1w = nf(); const float* l1b = nf();
            const float* l2w = nf(); const float* l2b = nf();

            float* xi = cur + (size_t)PREF * C;
            float* xo = alt + (size_t)PREF * C;

            // --- depthwise branch: rinv -> fused norm/conv/residual ---
            rms_rinv_kernel<<<(unsigned)(T / 8), 256, 0, stream>>>(xi, rinv, C, T);
            {
                const long n = T * (long)C;
                dwconv_block_kernel<<<(unsigned)((n + 255) / 256), 256, 0, stream>>>(
                    xi, rinv, cw, cb, cc, nw, gm, xo, C, T);
            }
            { float* t2 = cur; cur = alt; alt = t2; }
            xi = cur + (size_t)PREF * C; xo = alt + (size_t)PREF * C;

            // --- FFN branch: rinv + bf16 weights -> GEMM1(GELU) -> GEMM2(res) ---
            rms_rinv_kernel<<<(unsigned)(T / 8), 256, 0, stream>>>(xi, rinv, C, T);
            {
                const long nw1 = (long)C4 * C;
                f32_to_bf16_kernel<<<(unsigned)((nw1 + 255) / 256), 256, 0, stream>>>(
                    l1w, wb1, nw1);
                f32_to_bf16_kernel<<<(unsigned)((nw1 + 255) / 256), 256, 0, stream>>>(
                    l2w, wb2, nw1);
                dim3 g1((unsigned)(T / 128), (unsigned)((C4 + 63) / 64));
                gemm_wmma_kernel<MODE_GELU_BF16, float>
                    <<<g1, 256, 0, stream>>>(xi, (long)C, wb1, l1b, rinv, fnw,
                                             nullptr, nullptr, (void*)h1, (long)C4,
                                             (int)T, C4, C);
                dim3 g2((unsigned)(T / 128), (unsigned)((C + 63) / 64));
                gemm_wmma_kernel<MODE_RES, __bf16>
                    <<<g2, 256, 0, stream>>>(h1, (long)C4, wb2, l2b, nullptr, nullptr,
                                             xi, fgm, (void*)xo, (long)C,
                                             (int)T, C, C4);
            }
            { float* t2 = cur; cur = alt; alt = t2; }
        }

        if (s < 6) { // strided downsample conv as a free-im2col WMMA GEMM
            const float* dw = nf(); const float* db = nf(); const float* dc = nf();
            const int r = RAT[s], Kk = 2 * r, ctx = r;
            const int C2 = 2 * C;
            const long T2 = T / r;
            float* xi = cur + (size_t)PREF * C;
            float* xo = alt + (size_t)PREF * C2;

            fill_cache_rows_kernel<<<(unsigned)((C * ctx + 255) / 256), 256, 0, stream>>>(
                dc, xi - (size_t)ctx * C, C, ctx);
            {
                const long n = (long)C2 * C * Kk;
                reorder_w_kernel<<<(unsigned)((n + 255) / 256), 256, 0, stream>>>(
                    dw, wbf, C2, C, Kk);
            }
            dim3 g((unsigned)(T2 / 128), (unsigned)((C2 + 63) / 64));
            gemm_wmma_kernel<MODE_PLAIN, float>
                <<<g, 256, 0, stream>>>(xi - (size_t)ctx * C, (long)r * C, wbf, db,
                                        nullptr, nullptr, nullptr, nullptr,
                                        (void*)xo, (long)C2, (int)T2, C2, Kk * C);
            { float* t2 = cur; cur = alt; alt = t2; }
            T = T2;
        }
    }

    { // head: 2048 -> 128, k=7, causal; then transpose to (C, T)
        const float* hw = nf(); const float* hb = nf(); const float* hc = nf();
        const int Cf = 2048, Kk = 7, ctx = 6, No = 128;
        float* xi = cur + (size_t)PREF * Cf;
        fill_cache_rows_kernel<<<(unsigned)((Cf * ctx + 255) / 256), 256, 0, stream>>>(
            hc, xi - (size_t)ctx * Cf, Cf, ctx);
        {
            const long n = (long)No * Cf * Kk;
            reorder_w_kernel<<<(unsigned)((n + 255) / 256), 256, 0, stream>>>(
                hw, wbf, No, Cf, Kk);
        }
        dim3 g((unsigned)(T / 128), (unsigned)((No + 63) / 64));
        gemm_wmma_kernel<MODE_PLAIN, float>
            <<<g, 256, 0, stream>>>(xi - (size_t)ctx * Cf, (long)Cf, wbf, hb,
                                    nullptr, nullptr, nullptr, nullptr,
                                    (void*)htmp, (long)No, (int)T, No, Kk * Cf);
        transpose128_kernel<<<64, 256, 0, stream>>>(htmp, (float*)d_out);
    }
}
